// Seq2seqPredictor_18992345383425
// MI455X (gfx1250) — matrix operand order repaired and checked by reading, not compiled, not run on hardware
//
#include <hip/hip_runtime.h>
#include <cstddef>

// ---------------- constants (fixed by the reference) ----------------
#define PAD 0
#define BOS 1
#define EOS 2
#define VOCAB 50000
#define HID 512
#define BATCH 32
#define FIXLEN 32
#define KBEAM 8
#define BK (BATCH * KBEAM)          // 256
#define BANW ((VOCAB + 31) / 32)    // 1563 words per row
#define NEGF (-__builtin_inff())
#define MT 16                       // M-tiles per wave in the logits GEMM (256/16)

typedef __bf16 bf16;
typedef __attribute__((ext_vector_type(16))) __bf16 v16bf;
typedef __attribute__((ext_vector_type(8)))  __bf16 v8bf;
typedef __attribute__((ext_vector_type(8)))  float  v8f;

static __device__ __forceinline__ bf16 f2bf(float f) {
    union { float f; unsigned u; } x{f};
    unsigned r = x.u + 0x7fffu + ((x.u >> 16) & 1u);   // round-to-nearest-even
    unsigned short h = (unsigned short)(r >> 16);
    return __builtin_bit_cast(bf16, h);
}

// ---------------- fragment loads (bf16, 16x16x32) ----------------
// A (MxK row-major): lane holds row m=lane&15; khalf=lane>>4.
//   VGPR0-3 -> K = khalf*8 + 0..7 ; VGPR4-7 -> K = 16 + khalf*8 + 0..7
static __device__ __forceinline__ v16bf load_fragA(const bf16* A, int K, int m, int kb, int khalf) {
    const bf16* p = A + (size_t)m * K + kb + khalf * 8;
    v8bf lo = *reinterpret_cast<const v8bf*>(p);
    v8bf hi = *reinterpret_cast<const v8bf*>(p + 16);
    union { v16bf v; v8bf h[2]; } u;
    u.h[0] = lo; u.h[1] = hi;
    return u.v;
}
// B (KxN) supplied TRANSPOSED as BT (NxK row-major): lane holds column n=lane&15,
//   VGPR0-3 -> K = khalf*8 + 0..7 ; VGPR4-7 -> K = 16 + khalf*8 + 0..7 (same as A).
static __device__ __forceinline__ v16bf load_fragB(const bf16* BT, int K, int n, int kb, int khalf) {
    const bf16* p = BT + (size_t)n * K + kb + khalf * 8;
    v8bf lo = *reinterpret_cast<const v8bf*>(p);
    v8bf hi = *reinterpret_cast<const v8bf*>(p + 16);
    union { v16bf v; v8bf h[2]; } u;
    u.h[0] = lo; u.h[1] = hi;
    return u.v;
}

// ---------------- generic WMMA GEMM: C = act(A@B [+ A2@B2]) ----------------
// epi: 0 = store f32, 1 = tanh -> f32, 2 = tanh -> bf16
__global__ void gemm_bf16_k(const bf16* __restrict__ A,  const bf16* __restrict__ BT,
                            const bf16* __restrict__ A2, const bf16* __restrict__ B2T,
                            float* __restrict__ Cf, bf16* __restrict__ Cb,
                            int M, int N, int K, int epi)
{
    int wave = threadIdx.x >> 5;
    int lane = threadIdx.x & 31;
    int tileN = (blockIdx.x * 4 + wave) * 16;
    int tileM = blockIdx.y * 16;
    if (tileN >= N) return;                 // wave-uniform: EXEC stays all-ones
    int mrow  = tileM + (lane & 15);
    int ncol  = tileN + (lane & 15);
    int khalf = lane >> 4;

    v8f acc = {0.f, 0.f, 0.f, 0.f, 0.f, 0.f, 0.f, 0.f};
    for (int kb = 0; kb < K; kb += 32) {
        v16bf a = load_fragA(A,  K, mrow, kb, khalf);
        v16bf b = load_fragB(BT, K, ncol, kb, khalf);
        acc = __builtin_amdgcn_wmma_f32_16x16x32_bf16(false, a, false, b, (short)0, acc, false, false);
    }
    if (A2) {
        for (int kb = 0; kb < K; kb += 32) {
            v16bf a = load_fragA(A2,  K, mrow, kb, khalf);
            v16bf b = load_fragB(B2T, K, ncol, kb, khalf);
            acc = __builtin_amdgcn_wmma_f32_16x16x32_bf16(false, a, false, b, (short)0, acc, false, false);
        }
    }
#pragma unroll
    for (int r = 0; r < 8; ++r) {
        int m = tileM + r + 8 * khalf;
        int n = tileN + (lane & 15);
        float v = acc[r];
        if (epi >= 1) v = tanhf(v);
        if (epi == 2) Cb[(size_t)m * N + n] = f2bf(v);
        else          Cf[(size_t)m * N + n] = v;
    }
}

// ---------------- M-blocked WMMA GEMM for logits (M fixed = 256) ----------------
// One wave owns a 16-wide N-tile and ALL 16 M-tiles: each B fragment is loaded
// once and reused 16x in registers -> B streams from L2 exactly once per step.
__global__ void gemm_logits_k(const bf16* __restrict__ A, const bf16* __restrict__ BT,
                              float* __restrict__ C, int N, int K)
{
    int wave = threadIdx.x >> 5;
    int lane = threadIdx.x & 31;
    int tileN = (blockIdx.x * 4 + wave) * 16;
    if (tileN >= N) return;                 // wave-uniform
    int khalf = lane >> 4;
    int ncol  = tileN + (lane & 15);
    int mlane = lane & 15;

    v8f acc[MT];
#pragma unroll
    for (int t = 0; t < MT; ++t) acc[t] = {0.f, 0.f, 0.f, 0.f, 0.f, 0.f, 0.f, 0.f};

    for (int kb = 0; kb < K; kb += 32) {
        v16bf b = load_fragB(BT, K, ncol, kb, khalf);
#pragma unroll
        for (int t = 0; t < MT; ++t) {
            v16bf a = load_fragA(A, K, t * 16 + mlane, kb, khalf);
            acc[t] = __builtin_amdgcn_wmma_f32_16x16x32_bf16(false, a, false, b, (short)0, acc[t], false, false);
        }
    }
#pragma unroll
    for (int t = 0; t < MT; ++t) {
#pragma unroll
        for (int r = 0; r < 8; ++r) {
            int m = t * 16 + r + 8 * khalf;
            int n = tileN + (lane & 15);
            C[(size_t)m * N + n] = acc[t][r];
        }
    }
}

// ---------------- helpers ----------------
// f32 (K x N) -> bf16 transposed (N x K)
__global__ void transpose_bf16_k(const float* __restrict__ in, bf16* __restrict__ out, int Kd, int Nd)
{
    int idx = blockIdx.x * 256 + threadIdx.x;
    int total = Kd * Nd;
    if (idx >= total) return;
    int kk = idx / Nd;
    int n  = idx - kk * Nd;
    out[(size_t)n * Kd + kk] = f2bf(in[idx]);
}

// gather embedding rows -> bf16
__global__ void embed_rows_k(const float* __restrict__ E, const int* __restrict__ ids, bf16* __restrict__ out)
{
    int row = blockIdx.x;
    const float* src = E + (size_t)ids[row] * HID;
    for (int h = threadIdx.x; h < HID; h += blockDim.x)
        out[(size_t)row * HID + h] = f2bf(src[h]);
}

// mean over valid positions / seq_len -> bf16  (context already tanh'ed)
__global__ void mean_k(const float* __restrict__ context, const int* __restrict__ source,
                       const int* __restrict__ seq_len, bf16* __restrict__ mean)
{
    int b = blockIdx.x, h = threadIdx.x;            // blockDim = HID
    float s = 0.f;
    for (int t = 0; t < FIXLEN; ++t)
        if (source[b * FIXLEN + t] != PAD)
            s += context[((size_t)(b * FIXLEN + t)) * HID + h];
    mean[(size_t)b * HID + h] = f2bf(s / (float)seq_len[b]);
}

// replicate 32-row state to 256 beam rows, bf16
__global__ void repstate_k(const float* __restrict__ st, bf16* __restrict__ out)
{
    int row = blockIdx.x, h = threadIdx.x;          // grid 256, blockDim = HID
    out[(size_t)row * HID + h] = f2bf(st[(size_t)(row >> 3) * HID + h]);
}

// attention + residual: s = h + ctx, stored bf16
__global__ void attn_k(const float* __restrict__ hrow, const float* __restrict__ context,
                       const int* __restrict__ source, bf16* __restrict__ s_bf)
{
    int row = blockIdx.x, b = row >> 3, tid = threadIdx.x;   // blockDim 256
    __shared__ float hl[HID];
    __shared__ float w[FIXLEN];
    for (int h = tid; h < HID; h += blockDim.x) hl[h] = hrow[(size_t)row * HID + h];
    __syncthreads();
    if (tid < FIXLEN) {
        float a = 0.f;
        const float* cr = context + ((size_t)(b * FIXLEN + tid)) * HID;
        for (int h = 0; h < HID; ++h) a += hl[h] * cr[h];
        if (source[b * FIXLEN + tid] == PAD) a = -1e9f;
        w[tid] = a;
    }
    __syncthreads();
    if (tid == 0) {
        float mx = w[0];
        for (int t = 1; t < FIXLEN; ++t) mx = fmaxf(mx, w[t]);
        float s = 0.f;
        for (int t = 0; t < FIXLEN; ++t) { w[t] = expf(w[t] - mx); s += w[t]; }
        float inv = 1.f / s;
        for (int t = 0; t < FIXLEN; ++t) w[t] *= inv;
    }
    __syncthreads();
    for (int h = tid; h < HID; h += blockDim.x) {
        float c = 0.f;
        for (int t = 0; t < FIXLEN; ++t)
            c += w[t] * context[((size_t)(b * FIXLEN + t)) * HID + h];
        s_bf[(size_t)row * HID + h] = f2bf(hl[h] + c);
    }
}

// per-row max + log-sum-exp over logits
__global__ void rowstats_k(const float* __restrict__ logits, float* __restrict__ rmax, float* __restrict__ rlse)
{
    int row = blockIdx.x, tid = threadIdx.x;        // blockDim 256
    __shared__ float sm[256];
    float mx = NEGF;
    for (int v = tid; v < VOCAB; v += 256) mx = fmaxf(mx, logits[(size_t)row * VOCAB + v]);
    sm[tid] = mx; __syncthreads();
    for (int s = 128; s > 0; s >>= 1) { if (tid < s) sm[tid] = fmaxf(sm[tid], sm[tid + s]); __syncthreads(); }
    mx = sm[0]; __syncthreads();
    float sum = 0.f;
    for (int v = tid; v < VOCAB; v += 256) sum += expf(logits[(size_t)row * VOCAB + v] - mx);
    sm[tid] = sum; __syncthreads();
    for (int s = 128; s > 0; s >>= 1) { if (tid < s) sm[tid] += sm[tid + s]; __syncthreads(); }
    if (tid == 0) { rmax[row] = mx; rlse[row] = logf(sm[0]); }
}

// ---------------- masks / init ----------------
__global__ void build_masks_k(const int* __restrict__ source, const int* __restrict__ seq_len,
                              unsigned char* __restrict__ sentpat, unsigned char* __restrict__ known,
                              int* __restrict__ multi_any)
{
    int tid = threadIdx.x;                          // one block, 1024 threads
    for (int idx = tid; idx < BATCH * FIXLEN * FIXLEN; idx += blockDim.x) {
        int b = idx / (FIXLEN * FIXLEN);
        int r = idx - b * (FIXLEN * FIXLEN);
        int i = r / FIXLEN, j = r - i * FIXLEN;
        int si_ = source[b * FIXLEN + i], sj = source[b * FIXLEN + j];
        sentpat[idx] = (si_ == sj && si_ != PAD && sj != PAD) ? 1 : 0;
    }
    __syncthreads();
    for (int idx = tid; idx < BATCH * FIXLEN; idx += blockDim.x) {
        int b = idx / FIXLEN, i = idx - b * FIXLEN;
        int c = 0;
        for (int j = 0; j < i; ++j) c += sentpat[(b * FIXLEN + i) * FIXLEN + j];
        unsigned char kn = (c > 0) ? 1 : 0;
        if (i == seq_len[b] - 1) kn = 1;
        known[idx] = kn;
    }
    for (int i = tid; i < FIXLEN; i += blockDim.x) {
        int any = 0;
        for (int b = 0; b < BATCH; ++b) {
            int c = 0;
            for (int j = 0; j < FIXLEN; ++j) c += sentpat[(b * FIXLEN + i) * FIXLEN + j];
            if (c > 1) any = 1;
        }
        multi_any[i] = any;
    }
}

__global__ void init_k(unsigned* __restrict__ ban, int* __restrict__ outbuf,
                       float* __restrict__ scores, int* __restrict__ input_var,
                       const int* __restrict__ seq_len)
{
    size_t idx = (size_t)blockIdx.x * blockDim.x + threadIdx.x;
    if (idx < (size_t)BK * BANW) {
        int w = (int)(idx % BANW);
        ban[idx] = (w == 0) ? 0x7u : 0u;            // PAD, BOS, EOS banned
    }
    if (idx < (size_t)BATCH * KBEAM * FIXLEN) {
        int t = (int)(idx % FIXLEN);
        int b = (int)(idx / (KBEAM * FIXLEN));
        outbuf[idx] = (t == seq_len[b] - 1) ? EOS : PAD;
    }
    if (idx < BK) {
        scores[idx] = ((idx % KBEAM) == 0) ? 0.f : NEGF;
        input_var[idx] = BOS;
    }
}

// ---------------- fused score + top-k + beam bookkeeping ----------------
__global__ void topk_update_k(const float* __restrict__ logits, const float* __restrict__ rmax,
                              const float* __restrict__ rlse, float* __restrict__ scores,
                              const int* __restrict__ seq_len, const unsigned char* __restrict__ known,
                              const int* __restrict__ out_old, const unsigned* __restrict__ ban,
                              int* __restrict__ combine, int* __restrict__ symbol,
                              int* __restrict__ kidx_out, int* __restrict__ input_var, int step)
{
    int b = blockIdx.x, tid = threadIdx.x;          // blockDim 256
    if (seq_len[b] <= step) {                       // whole batch finished: freeze
        if (tid < KBEAM) {
            combine[b * KBEAM + tid]  = b * KBEAM + tid;
            kidx_out[b * KBEAM + tid] = tid;
            symbol[b * KBEAM + tid]   = PAD;
            input_var[b * KBEAM + tid] = PAD;
        }
        return;
    }
    __shared__ float sOld[KBEAM], sMx[KBEAM], sLse[KBEAM];
    __shared__ int   sAllowed[KBEAM];
    __shared__ float sv[256 * 8];
    __shared__ int   si[256 * 8];
    __shared__ float rv[256];
    __shared__ int   rp[256];
    __shared__ float wval[KBEAM];
    __shared__ int   wcand[KBEAM];

    if (tid < KBEAM) {
        sOld[tid] = scores[b * KBEAM + tid];
        sMx[tid]  = rmax[b * KBEAM + tid];
        sLse[tid] = rlse[b * KBEAM + tid];
        sAllowed[tid] = out_old[(b * KBEAM + tid) * FIXLEN + step];
    }
    __syncthreads();
    bool kn = known[b * FIXLEN + step] != 0;

    float tv[8]; int ti[8];
#pragma unroll
    for (int q = 0; q < 8; ++q) { tv[q] = NEGF; ti[q] = 0x7fffffff; }

    for (int c = tid; c < KBEAM * VOCAB; c += 256) {
        int kb = c / VOCAB;
        int v  = c - kb * VOCAB;
        float sc;
        if (kn) {
            sc = (v == sAllowed[kb])
                 ? sOld[kb] + (logits[(size_t)(b * KBEAM + kb) * VOCAB + v] - sMx[kb] - sLse[kb])
                 : NEGF;
        } else {
            bool banned = (ban[(size_t)(b * KBEAM + kb) * BANW + (v >> 5)] >> (v & 31)) & 1u;
            sc = banned ? NEGF
                        : sOld[kb] + (logits[(size_t)(b * KBEAM + kb) * VOCAB + v] - sMx[kb] - sLse[kb]);
        }
        if (sc > tv[7] || (sc == tv[7] && c < ti[7])) {
            tv[7] = sc; ti[7] = c;
#pragma unroll
            for (int q = 7; q > 0; --q) {
                if (tv[q] > tv[q - 1] || (tv[q] == tv[q - 1] && ti[q] < ti[q - 1])) {
                    float fv = tv[q]; tv[q] = tv[q - 1]; tv[q - 1] = fv;
                    int   iv = ti[q]; ti[q] = ti[q - 1]; ti[q - 1] = iv;
                } else break;
            }
        }
    }
#pragma unroll
    for (int q = 0; q < 8; ++q) { sv[tid * 8 + q] = tv[q]; si[tid * 8 + q] = ti[q]; }
    __syncthreads();

    for (int sel = 0; sel < KBEAM; ++sel) {
        float best = NEGF; int bp = tid * 8;
#pragma unroll
        for (int q = 0; q < 8; ++q) {
            float vq = sv[tid * 8 + q]; int iq = si[tid * 8 + q];
            if (vq > best || (vq == best && iq < si[bp])) { best = vq; bp = tid * 8 + q; }
        }
        rv[tid] = best; rp[tid] = bp;
        __syncthreads();
        for (int s = 128; s > 0; s >>= 1) {
            if (tid < s) {
                float v2 = rv[tid + s]; int p2 = rp[tid + s];
                if (v2 > rv[tid] || (v2 == rv[tid] && si[p2] < si[rp[tid]])) { rv[tid] = v2; rp[tid] = p2; }
            }
            __syncthreads();
        }
        if (tid == 0) {
            wval[sel] = rv[0];
            wcand[sel] = si[rp[0]];
            sv[rp[0]] = NEGF;                       // consume winner
        }
        __syncthreads();
    }
    if (tid < KBEAM) {
        int cand = wcand[tid];
        int kb = cand / VOCAB, v = cand - kb * VOCAB;
        scores[b * KBEAM + tid]   = wval[tid];
        combine[b * KBEAM + tid]  = b * KBEAM + kb;
        kidx_out[b * KBEAM + tid] = kb;
        symbol[b * KBEAM + tid]   = v;
        input_var[b * KBEAM + tid] = v;
    }
}

// gather new beam state (bf16 for next GEMM) + permuted ban with new token banned
__global__ void reorder_state_ban_k(const float* __restrict__ h, bf16* __restrict__ state_bf,
                                    const unsigned* __restrict__ ban_in, unsigned* __restrict__ ban_out,
                                    const int* __restrict__ combine, const int* __restrict__ symbol)
{
    int row = blockIdx.x, tid = threadIdx.x;        // grid 256, blockDim 256
    int src = combine[row];
    int tok = symbol[row];
    for (int hh = tid; hh < HID; hh += blockDim.x)
        state_bf[(size_t)row * HID + hh] = f2bf(h[(size_t)src * HID + hh]);
    for (int w = tid; w < BANW; w += blockDim.x) {
        unsigned word = ban_in[(size_t)src * BANW + w];
        if (w == (tok >> 5)) word |= (1u << (tok & 31));
        ban_out[(size_t)row * BANW + w] = word;
    }
}

// permute output rows by k_idx then scatter symbol (out_multi / out_simple)
__global__ void reorder_output_k(const int* __restrict__ out_in, int* __restrict__ out_out,
                                 const int* __restrict__ kidx, const int* __restrict__ symbol,
                                 const unsigned char* __restrict__ sentpat,
                                 const int* __restrict__ multi_any, int step)
{
    int b = blockIdx.x, tid = threadIdx.x;          // blockDim 256 = 8 beams x 32 pos
    int j = tid >> 5, t = tid & 31;
    int kb = kidx[b * KBEAM + j];
    int val = out_in[(b * KBEAM + kb) * FIXLEN + t];
    int sym = symbol[b * KBEAM + j];
    if (multi_any[step]) {
        if (sentpat[(b * FIXLEN + step) * FIXLEN + t]) val = sym;
    } else if (t == step) {
        val = sym;
    }
    out_out[(b * KBEAM + j) * FIXLEN + t] = val;
}

__global__ void finalize_k(const int* __restrict__ outbuf, const float* __restrict__ scores,
                           float* __restrict__ d_out)
{
    int idx = blockIdx.x * 256 + threadIdx.x;
    if (idx < BATCH * KBEAM * FIXLEN) d_out[idx] = (float)outbuf[idx];
    else if (idx < BATCH * KBEAM * FIXLEN + BK) d_out[idx] = scores[idx - BATCH * KBEAM * FIXLEN];
}

// ---------------- host ----------------
struct Carve {
    char* p;
    template <typename T> T* take(size_t n) {
        T* r = (T*)p;
        size_t bytes = (n * sizeof(T) + 255) & ~(size_t)255;
        p += bytes;
        return r;
    }
};

extern "C" void kernel_launch(void* const* d_in, const int* in_sizes, int n_in,
                              void* d_out, int out_size, void* d_ws, size_t ws_size,
                              hipStream_t stream)
{
    (void)in_sizes; (void)n_in; (void)out_size; (void)ws_size;
    const int*   source  = (const int*)  d_in[0];
    const int*   seq_len = (const int*)  d_in[1];
    const float* E       = (const float*)d_in[3];
    const float* W_enc   = (const float*)d_in[4];
    const float* W_s     = (const float*)d_in[5];
    const float* Wx      = (const float*)d_in[6];
    const float* Wh      = (const float*)d_in[7];
    const float* Wf      = (const float*)d_in[8];
    const float* W_out   = (const float*)d_in[9];
    float* out_f = (float*)d_out;

    Carve cv{(char*)d_ws};
    bf16* WencT = cv.take<bf16>((size_t)HID * HID);
    bf16* WsT   = cv.take<bf16>((size_t)HID * HID);
    bf16* WxT   = cv.take<bf16>((size_t)HID * HID);
    bf16* WhT   = cv.take<bf16>((size_t)HID * HID);
    bf16* WfT   = cv.take<bf16>((size_t)HID * HID);
    bf16* WoutT = cv.take<bf16>((size_t)VOCAB * HID);
    bf16*  xsrc_bf   = cv.take<bf16>((size_t)BATCH * FIXLEN * HID);
    float* context   = cv.take<float>((size_t)BATCH * FIXLEN * HID);
    bf16*  mean_bf   = cv.take<bf16>((size_t)BATCH * HID);
    float* stateSm   = cv.take<float>((size_t)BATCH * HID);
    bf16*  state_bf  = cv.take<bf16>((size_t)BK * HID);
    bf16*  x_bf      = cv.take<bf16>((size_t)BK * HID);
    float* hbuf      = cv.take<float>((size_t)BK * HID);
    bf16*  s_bf      = cv.take<bf16>((size_t)BK * HID);
    bf16*  fh_bf     = cv.take<bf16>((size_t)BK * HID);
    float* logits    = cv.take<float>((size_t)BK * VOCAB);
    float* rmax      = cv.take<float>(BK);
    float* rlse      = cv.take<float>(BK);
    float* scores    = cv.take<float>(BK);
    unsigned* banA   = cv.take<unsigned>((size_t)BK * BANW);
    unsigned* banB   = cv.take<unsigned>((size_t)BK * BANW);
    int* outA        = cv.take<int>((size_t)BATCH * KBEAM * FIXLEN);
    int* outB        = cv.take<int>((size_t)BATCH * KBEAM * FIXLEN);
    unsigned char* sentpat = cv.take<unsigned char>((size_t)BATCH * FIXLEN * FIXLEN);
    unsigned char* known   = cv.take<unsigned char>((size_t)BATCH * FIXLEN);
    int* multi_any   = cv.take<int>(FIXLEN);
    int* combine     = cv.take<int>(BK);
    int* symbol      = cv.take<int>(BK);
    int* kidx        = cv.take<int>(BK);
    int* input_var   = cv.take<int>(BK);

    auto gemm = [&](const bf16* A, const bf16* BT, const bf16* A2, const bf16* B2T,
                    float* Cf, bf16* Cb, int M, int N, int K, int epi) {
        dim3 grid((N / 16 + 3) / 4, M / 16);
        gemm_bf16_k<<<grid, 128, 0, stream>>>(A, BT, A2, B2T, Cf, Cb, M, N, K, epi);
    };
    auto transp = [&](const float* in, bf16* out, int Kd, int Nd) {
        int total = Kd * Nd;
        transpose_bf16_k<<<(total + 255) / 256, 256, 0, stream>>>(in, out, Kd, Nd);
    };

    // ---- one-time init / encoder ----
    init_k<<<((size_t)BK * BANW + 255) / 256, 256, 0, stream>>>(banA, outA, scores, input_var, seq_len);
    build_masks_k<<<1, 1024, 0, stream>>>(source, seq_len, sentpat, known, multi_any);
    transp(W_enc, WencT, HID, HID);
    transp(W_s,   WsT,   HID, HID);
    transp(Wx,    WxT,   HID, HID);
    transp(Wh,    WhT,   HID, HID);
    transp(Wf,    WfT,   HID, HID);
    transp(W_out, WoutT, HID, VOCAB);
    embed_rows_k<<<BATCH * FIXLEN, 256, 0, stream>>>(E, source, xsrc_bf);
    gemm(xsrc_bf, WencT, nullptr, nullptr, context, nullptr, BATCH * FIXLEN, HID, HID, 1); // tanh
    mean_k<<<BATCH, HID, 0, stream>>>(context, source, seq_len, mean_bf);
    gemm(mean_bf, WsT, nullptr, nullptr, stateSm, nullptr, BATCH, HID, HID, 1);            // tanh
    repstate_k<<<BK, HID, 0, stream>>>(stateSm, state_bf);

    unsigned* ban_cur = banA; unsigned* ban_nxt = banB;
    int* out_cur = outA;      int* out_nxt = outB;

    // ---- decode loop (unrolled on host; ~9 launches/step) ----
    for (int i = 0; i < FIXLEN; ++i) {
        embed_rows_k<<<BK, 256, 0, stream>>>(E, input_var, x_bf);
        // h = tanh(x@Wx + state@Wh)
        gemm(x_bf, WxT, state_bf, WhT, hbuf, nullptr, BK, HID, HID, 1);
        attn_k<<<BK, 256, 0, stream>>>(hbuf, context, source, s_bf);
        // fh = tanh((h+ctx)@Wf), bf16 for the big GEMM
        gemm(s_bf, WfT, nullptr, nullptr, nullptr, fh_bf, BK, HID, HID, 2);
        // logits = fh @ W_out : M-blocked, B fragment reused 16x in registers,
        // W_out bf16 (51 MB) streams from L2 exactly once per step.
        {
            dim3 grid((VOCAB / 16 + 3) / 4, 1);
            gemm_logits_k<<<grid, 128, 0, stream>>>(fh_bf, WoutT, logits, VOCAB, HID);
        }
        rowstats_k<<<BK, 256, 0, stream>>>(logits, rmax, rlse);
        topk_update_k<<<BATCH, 256, 0, stream>>>(logits, rmax, rlse, scores, seq_len, known,
                                                 out_cur, ban_cur, combine, symbol, kidx, input_var, i);
        reorder_state_ban_k<<<BK, 256, 0, stream>>>(hbuf, state_bf, ban_cur, ban_nxt, combine, symbol);
        reorder_output_k<<<BATCH, 256, 0, stream>>>(out_cur, out_nxt, kidx, symbol, sentpat, multi_any, i);
        { unsigned* t = ban_cur; ban_cur = ban_nxt; ban_nxt = t; }
        { int* t = out_cur; out_cur = out_nxt; out_nxt = t; }
    }

    finalize_k<<<(BATCH * KBEAM * FIXLEN + BK + 255) / 256, 256, 0, stream>>>(out_cur, scores, out_f);
}
